// FactoredTransformerBlock_30717606101352
// MI455X (gfx1250) — matrix operand order repaired and checked by reading, not compiled, not run on hardware
//
#include <hip/hip_runtime.h>

// ---------------------------------------------------------------------------
// CDNA5 (gfx1250) implementation of the factored transformer block.
// All GEMMs and both attention cores use v_wmma_f32_16x16x32_bf16
// (bf16 inputs, fp32 accumulate). wave32 everywhere.
// GEMM stages its A panel through LDS with global_load_async_to_lds_b128
// (ASYNCcnt-tracked, double buffered) and uses 2x2 register blocking.
// ---------------------------------------------------------------------------

typedef __bf16 bf16_t;
typedef __attribute__((ext_vector_type(16))) __bf16 v16bf;
typedef __attribute__((ext_vector_type(8)))  float  v8f;

#define D_MODEL   256
#define NUM_HEADS 8
#define HEAD_DIM  32
#define FF_DIM    1024
#define HIDDEN_C  128
#define BIAS_E    16
#define B_SZ      8
#define T_SZ      64
#define NA_SZ     32
#define MTOK      (B_SZ * T_SZ * NA_SZ)   // 16384 tokens
#define LN_EPS    1e-5f

__device__ __forceinline__ v8f v8f_zero() {
  v8f z = {0.f, 0.f, 0.f, 0.f, 0.f, 0.f, 0.f, 0.f};
  return z;
}

// A-matrix fragment (16x32 bf16, M x K). ISA layout:
// lanes 0-15: row M=lane, K = {0..7, 16..23}; lanes 16-31: row M=lane-16, K = {8..15, 24..31}
__device__ __forceinline__ v16bf load_frag_a(const bf16_t* __restrict__ base, int ld) {
  const int lane = threadIdx.x & 31;
  const int half = lane >> 4;
  const int r    = lane & 15;
  const bf16_t* p = base + (size_t)r * ld + half * 8;
  union { uint4 u[2]; v16bf v; } u;
  u.u[0] = *reinterpret_cast<const uint4*>(p);
  u.u[1] = *reinterpret_cast<const uint4*>(p + 16);
  return u.v;
}

// B-matrix fragment (32x16 bf16, K x N), sourced from a [N, K] (transposed) array.
// lanes 0-15: col N=lane, K = {0..7, 16..23}; lanes 16-31: col N=lane-16, K = {8..15, 24..31}
__device__ __forceinline__ v16bf load_frag_b(const bf16_t* __restrict__ base, int ld) {
  const int lane = threadIdx.x & 31;
  const int half = lane >> 4;
  const int c    = lane & 15;
  const bf16_t* p = base + (size_t)c * ld + half * 8;
  union { uint4 u[2]; v16bf v; } u;
  u.u[0] = *reinterpret_cast<const uint4*>(p);
  u.u[1] = *reinterpret_cast<const uint4*>(p + 16);
  return u.v;
}

__device__ __forceinline__ v8f wmma_bf16(v16bf a, v16bf b, v8f c) {
  return __builtin_amdgcn_wmma_f32_16x16x32_bf16(false, a, false, b, (short)0, c,
                                                 false, false);
}

// ---------------------------------------------------------------------------
// WMMA GEMM: C[M,N] = A[M,K] @ W[K,N] (+bias), W given transposed [N,K].
// mode 0: fp32 out; mode 1: relu -> bf16 out; mode 2: bf16 out.
// Block = 256 threads (8 waves, 2(M) x 4(N)); block tile 64x128; wave tile 32x32
// (2x2 register blocking). A panel (64x32 per K-step) is staged into LDS with
// double-buffered global_load_async_to_lds_b128, overlapped with WMMA compute.
// ---------------------------------------------------------------------------
__global__ void gemm_kernel(const bf16_t* __restrict__ A, const bf16_t* __restrict__ WT,
                            const float* __restrict__ bias, void* __restrict__ Cout,
                            int M, int N, int K, int mode) {
  const int wave  = threadIdx.x >> 5;
  const int lane  = threadIdx.x & 31;
  const int half  = lane >> 4;
  const int c     = lane & 15;
  const int waveM = wave >> 2;                     // 0..1
  const int waveN = wave & 3;                      // 0..3
  const int row0  = blockIdx.y * 64;
  const int col0  = blockIdx.x * 128 + waveN * 32;

  __shared__ __align__(16) bf16_t panelA[2][64 * 32];   // 2 x 4KB double buffer

  // Async fill: each of the 256 threads moves one 16B chunk of the 4KB panel.
  const int arow = threadIdx.x >> 2;               // 0..63
  const int acol = (threadIdx.x & 3) * 8;          // 0,8,16,24
  const uint64_t gbase = (uint64_t)(uintptr_t)A;
  const uint32_t goff_base =
      (uint32_t)(((size_t)(row0 + arow) * K + acol) * sizeof(bf16_t));
  uint32_t lds_dst[2];
  lds_dst[0] = (uint32_t)(size_t)&panelA[0][arow * 32 + acol];
  lds_dst[1] = (uint32_t)(size_t)&panelA[1][arow * 32 + acol];

  auto issueA = [&](int s) {
    const uint32_t go = goff_base + (uint32_t)(s * 64);  // k0 = s*32 bf16 = 64B
    asm volatile("global_load_async_to_lds_b128 %0, %1, %2"
                 :: "v"(lds_dst[s & 1]), "v"(go), "s"(gbase)
                 : "memory");
  };

  const int S = K / 32;
  issueA(0);
  issueA(1);

  const bf16_t* Bp0 = WT + (size_t)(col0 + 0)  * K;
  const bf16_t* Bp1 = WT + (size_t)(col0 + 16) * K;

  v8f acc00 = v8f_zero(), acc01 = v8f_zero();
  v8f acc10 = v8f_zero(), acc11 = v8f_zero();

  for (int s = 0; s < S; ++s) {
    // Own wave's oldest async panel complete (in-order), then publish to block.
    if (s == S - 1) asm volatile("s_wait_asynccnt 0x0" ::: "memory");
    else            asm volatile("s_wait_asynccnt 0x1" ::: "memory");
    __syncthreads();

    const bf16_t* ap = &panelA[s & 1][0];
    const int k0 = s * 32;
    __builtin_prefetch(Bp0 + k0 + 32, 0, 0);             // global_prefetch_b8
    v16bf a0 = load_frag_a(ap + (waveM * 32 + 0)  * 32, 32);
    v16bf a1 = load_frag_a(ap + (waveM * 32 + 16) * 32, 32);
    v16bf b0 = load_frag_b(Bp0 + k0, K);
    v16bf b1 = load_frag_b(Bp1 + k0, K);
    acc00 = wmma_bf16(a0, b0, acc00);
    acc01 = wmma_bf16(a0, b1, acc01);
    acc10 = wmma_bf16(a1, b0, acc10);
    acc11 = wmma_bf16(a1, b1, acc11);

    __syncthreads();                                      // all done reading buf
    if (s + 2 < S) issueA(s + 2);
  }

  auto store_tile = [&](v8f acc, int rbase, int cbase) {
    union { v8f v; float f[8]; } r;
    r.v = acc;
    const int coln = cbase + c;
    const float bv = bias ? bias[coln] : 0.f;
#pragma unroll
    for (int i = 0; i < 8; ++i) {
      const int row = rbase + half * 8 + i;
      float v = r.f[i] + bv;
      if (mode == 0) {
        reinterpret_cast<float*>(Cout)[(size_t)row * N + coln] = v;
      } else {
        if (mode == 1) v = fmaxf(v, 0.f);
        reinterpret_cast<bf16_t*>(Cout)[(size_t)row * N + coln] = (bf16_t)v;
      }
    }
  };
  store_tile(acc00, row0 + waveM * 32 + 0,  col0 + 0);
  store_tile(acc01, row0 + waveM * 32 + 0,  col0 + 16);
  store_tile(acc10, row0 + waveM * 32 + 16, col0 + 0);
  store_tile(acc11, row0 + waveM * 32 + 16, col0 + 16);
}

// ---------------------------------------------------------------------------
// Fused MHSA core for one (sequence, head).  NT16 = n/16 query strips.
// q/k/v: bf16 [nseq*n, 256], token-major; bias: fp32 [seq, (head), n, n].
// One wave per 16-query strip: scores (WMMA, K=32) -> softmax -> AV (WMMA).
// ---------------------------------------------------------------------------
template <int NT16>
__global__ void attn_kernel(const bf16_t* __restrict__ q, const bf16_t* __restrict__ kk,
                            const bf16_t* __restrict__ vv,
                            const float* __restrict__ bias, int bias_sseq, int bias_shead,
                            bf16_t* __restrict__ out, float scale) {
  constexpr int N = NT16 * 16;   // sequence length (64 or 32)
  const int seq  = blockIdx.x >> 3;
  const int h    = blockIdx.x & 7;
  const int wave = threadIdx.x >> 5;
  const int lane = threadIdx.x & 31;
  const int half = lane >> 4;
  const int c    = lane & 15;

  __shared__ __align__(16) bf16_t vT[HEAD_DIM][N + 8];     // padded: avoid bank conflicts
  __shared__ __align__(16) bf16_t alpha[NT16][16][N];

  // Cooperative transpose of V head-slice into LDS: vT[d][t]
  for (int idx = threadIdx.x; idx < HEAD_DIM * N; idx += blockDim.x) {
    const int t = idx >> 5;
    const int d = idx & 31;
    vT[d][t] = vv[((size_t)seq * N + t) * D_MODEL + h * HEAD_DIM + d];
  }
  __syncthreads();

  // Scores: S = Q(16x32) @ K^T(32x16 per tile)
  v16bf aq = load_frag_a(q + ((size_t)seq * N + wave * 16) * D_MODEL + h * HEAD_DIM, D_MODEL);
  float s[NT16][8];
#pragma unroll
  for (int ct = 0; ct < NT16; ++ct) {
    v16bf bk = load_frag_b(kk + ((size_t)seq * N + ct * 16) * D_MODEL + h * HEAD_DIM, D_MODEL);
    union { v8f v; float f[8]; } r;
    r.v = wmma_bf16(aq, bk, v8f_zero());
#pragma unroll
    for (int i = 0; i < 8; ++i) s[ct][i] = r.f[i];
  }

  // scale + additive bias
  const float* bp = bias + (size_t)seq * bias_sseq + (size_t)h * bias_shead;
#pragma unroll
  for (int ct = 0; ct < NT16; ++ct) {
#pragma unroll
    for (int i = 0; i < 8; ++i) {
      const int mq = wave * 16 + half * 8 + i;
      const int nk = ct * 16 + c;
      s[ct][i] = s[ct][i] * scale + bp[mq * N + nk];
    }
  }

  // Row softmax: row m lives in element i across the 16 lanes of this half.
#pragma unroll
  for (int i = 0; i < 8; ++i) {
    float m = s[0][i];
#pragma unroll
    for (int ct = 1; ct < NT16; ++ct) m = fmaxf(m, s[ct][i]);
#pragma unroll
    for (int off = 8; off >= 1; off >>= 1) m = fmaxf(m, __shfl_xor(m, off, 32));
    float sum = 0.f;
#pragma unroll
    for (int ct = 0; ct < NT16; ++ct) {
      const float e = __expf(s[ct][i] - m);
      s[ct][i] = e;
      sum += e;
    }
#pragma unroll
    for (int off = 8; off >= 1; off >>= 1) sum += __shfl_xor(sum, off, 32);
    const float inv = 1.f / sum;
#pragma unroll
    for (int ct = 0; ct < NT16; ++ct)
      alpha[wave][half * 8 + i][ct * 16 + c] = (bf16_t)(s[ct][i] * inv);
  }
  // alpha written & read by the same wave; LDS ops are in-order per wave.

  // AV: O(16x32) = alpha(16xN) @ V(Nx32), two 16-col tiles of head_dim
#pragma unroll
  for (int ot = 0; ot < 2; ++ot) {
    v8f acc = v8f_zero();
#pragma unroll
    for (int kt = 0; kt < N / 32; ++kt) {
      v16bf aa = load_frag_a(&alpha[wave][0][kt * 32], N);
      v16bf bv = load_frag_b(&vT[ot * 16][kt * 32], N + 8);
      acc = wmma_bf16(aa, bv, acc);
    }
    union { v8f v; float f[8]; } r;
    r.v = acc;
#pragma unroll
    for (int i = 0; i < 8; ++i) {
      const int mq = wave * 16 + half * 8 + i;
      out[((size_t)seq * N + mq) * D_MODEL + h * HEAD_DIM + ot * 16 + c] = (bf16_t)r.f[i];
    }
  }
}

// ---------------------------------------------------------------------------
// Fused residual add + LayerNorm.  One wave per 256-wide row (8 rows/block).
// permute!=0 remaps temporal row (b*NA+a)*T+t -> asset row (b*T+t)*NA+a.
// ---------------------------------------------------------------------------
__global__ void add_ln_kernel(const float* __restrict__ resid, const float* __restrict__ y,
                              const float* __restrict__ g, const float* __restrict__ bta,
                              bf16_t* __restrict__ out_bf, float* __restrict__ out_f,
                              int permute) {
  const int wave = threadIdx.x >> 5;
  const int lane = threadIdx.x & 31;
  const int row  = blockIdx.x * 8 + wave;

  const float* rp = resid + (size_t)row * D_MODEL + lane * 8;
  const float* yp = y     + (size_t)row * D_MODEL + lane * 8;
  float xv[8];
#pragma unroll
  for (int i = 0; i < 8; ++i) xv[i] = rp[i] + yp[i];

  float sm = 0.f, ss = 0.f;
#pragma unroll
  for (int i = 0; i < 8; ++i) { sm += xv[i]; ss += xv[i] * xv[i]; }
#pragma unroll
  for (int off = 16; off >= 1; off >>= 1) {
    sm += __shfl_xor(sm, off, 32);
    ss += __shfl_xor(ss, off, 32);
  }
  const float mean = sm * (1.f / D_MODEL);
  const float var  = ss * (1.f / D_MODEL) - mean * mean;
  const float inv  = rsqrtf(var + LN_EPS);

  int orow = row;
  if (permute) {
    const int b   = row / (NA_SZ * T_SZ);
    const int rem = row % (NA_SZ * T_SZ);
    const int a   = rem / T_SZ;
    const int t   = rem % T_SZ;
    orow = (b * T_SZ + t) * NA_SZ + a;
  }
#pragma unroll
  for (int i = 0; i < 8; ++i) {
    const int d = lane * 8 + i;
    const float v = (xv[i] - mean) * inv * g[d] + bta[d];
    if (out_bf) out_bf[(size_t)orow * D_MODEL + d] = (bf16_t)v;
    if (out_f)  out_f[(size_t)orow * D_MODEL + d]  = v;
  }
}

// x [b,t,a,d] fp32 -> temporal layout [(b*NA+a)*T+t, d] as bf16 + fp32 copy
__global__ void prep_x_kernel(const float* __restrict__ x,
                              bf16_t* __restrict__ xb, float* __restrict__ xf) {
  const int rt = blockIdx.x;            // temporal row
  const int d  = threadIdx.x;
  const int sq = rt / T_SZ, t = rt % T_SZ;
  const int b  = sq / NA_SZ, a = sq % NA_SZ;
  const size_t in = ((size_t)(b * T_SZ + t) * NA_SZ + a) * D_MODEL + d;
  const float v = x[in];
  xb[(size_t)rt * D_MODEL + d] = (bf16_t)v;
  xf[(size_t)rt * D_MODEL + d] = v;
}

// fp32 W[K,N] -> bf16 W^T[N,K]
__global__ void convert_wT_kernel(const float* __restrict__ w, bf16_t* __restrict__ wt,
                                  int K, int N) {
  const int idx = blockIdx.x * blockDim.x + threadIdx.x;
  const int k = idx / N, n = idx % N;
  wt[(size_t)n * K + k] = (bf16_t)w[idx];
}

// dyn_bias: bias[bt,h,j,l] = gamma * sum_e qvec[token(j),h,e] * beta[token(j),l,h,e]
// One block per token=(b*T+t)*NA+j; thread = (l,h). Pure-bandwidth beta stream.
__global__ void dyn_bias_kernel(const float* __restrict__ qvec, const float* __restrict__ beta,
                                const float* __restrict__ gamma, float* __restrict__ abias) {
  const int token = blockIdx.x;
  __shared__ float qs[NUM_HEADS * BIAS_E];
  if (threadIdx.x < NUM_HEADS * BIAS_E)
    qs[threadIdx.x] = qvec[(size_t)token * (NUM_HEADS * BIAS_E) + threadIdx.x];
  __syncthreads();

  const int l = threadIdx.x >> 3;
  const int h = threadIdx.x & 7;
  const float4* bp = reinterpret_cast<const float4*>(
      beta + ((size_t)token * NA_SZ + l) * (NUM_HEADS * BIAS_E) + h * BIAS_E);
  const float* qh = qs + h * BIAS_E;
  float s = 0.f;
#pragma unroll
  for (int j = 0; j < 4; ++j) {
    const float4 b4 = bp[j];
    s += b4.x * qh[4 * j + 0] + b4.y * qh[4 * j + 1] +
         b4.z * qh[4 * j + 2] + b4.w * qh[4 * j + 3];
  }
  const int bt = token >> 5;       // /NA
  const int jq = token & 31;       // %NA
  abias[((size_t)bt * NUM_HEADS + h) * (NA_SZ * NA_SZ) + jq * NA_SZ + l] = gamma[0] * s;
}

// ---------------------------------------------------------------------------
// Host orchestration
// ---------------------------------------------------------------------------
enum {
  IN_X = 0, IN_BETA, IN_GAMMA, IN_TMASK,
  T_QW, T_QB, T_KW, T_KB, T_VW, T_VB, T_OW, T_OB, T_LNG, T_LNB,        // t_attn
  TF_W1, TF_B1, TF_W2, TF_B2, TF_LNG, TF_LNB,                          // t_ffn
  A_QW, A_QB, A_KW, A_KB, A_VW, A_VB, A_OW, A_OB, A_LNG, A_LNB,        // a_attn
  AF_W1, AF_B1, AF_W2, AF_B2, AF_LNG, AF_LNB,                          // a_ffn
  Q_L1W, Q_L1B, Q_L2W                                                  // qmlp
};

extern "C" void kernel_launch(void* const* d_in, const int* in_sizes, int n_in,
                              void* d_out, int out_size, void* d_ws, size_t ws_size,
                              hipStream_t stream) {
  (void)in_sizes; (void)n_in; (void)out_size; (void)ws_size;
  char* ws = reinterpret_cast<char*>(d_ws);
  size_t off = 0;
  auto alloc = [&](size_t bytes) -> void* {
    off = (off + 255) & ~(size_t)255;
    void* p = ws + off;
    off += bytes;
    return p;
  };
  const size_t M = MTOK;

  // transposed bf16 weights
  bf16_t* WqT_t = (bf16_t*)alloc(256 * 256 * 2);
  bf16_t* WkT_t = (bf16_t*)alloc(256 * 256 * 2);
  bf16_t* WvT_t = (bf16_t*)alloc(256 * 256 * 2);
  bf16_t* WoT_t = (bf16_t*)alloc(256 * 256 * 2);
  bf16_t* W1T_t = (bf16_t*)alloc(256 * 1024 * 2);
  bf16_t* W2T_t = (bf16_t*)alloc(1024 * 256 * 2);
  bf16_t* WqT_a = (bf16_t*)alloc(256 * 256 * 2);
  bf16_t* WkT_a = (bf16_t*)alloc(256 * 256 * 2);
  bf16_t* WvT_a = (bf16_t*)alloc(256 * 256 * 2);
  bf16_t* WoT_a = (bf16_t*)alloc(256 * 256 * 2);
  bf16_t* W1T_a = (bf16_t*)alloc(256 * 1024 * 2);
  bf16_t* W2T_a = (bf16_t*)alloc(1024 * 256 * 2);
  bf16_t* L1T   = (bf16_t*)alloc(256 * 128 * 2);
  bf16_t* L2T   = (bf16_t*)alloc(128 * 128 * 2);

  // activations (ping-pong pairs: bf16 for GEMM input, fp32 residual)
  bf16_t* xa_bf = (bf16_t*)alloc(M * 256 * 2);
  float*  xa_f  = (float*) alloc(M * 256 * 4);
  bf16_t* xb_bf = (bf16_t*)alloc(M * 256 * 2);
  float*  xb_f  = (float*) alloc(M * 256 * 4);
  bf16_t* qb    = (bf16_t*)alloc(M * 256 * 2);
  bf16_t* kb    = (bf16_t*)alloc(M * 256 * 2);
  bf16_t* vb    = (bf16_t*)alloc(M * 256 * 2);
  bf16_t* ao    = (bf16_t*)alloc(M * 256 * 2);
  float*  gout  = (float*) alloc(M * 256 * 4);
  bf16_t* hbuf  = (bf16_t*)alloc(M * 1024 * 2);
  bf16_t* q1    = (bf16_t*)alloc(M * 128 * 2);
  float*  qvec  = (float*) alloc(M * 128 * 4);
  float*  abias = (float*) alloc((size_t)B_SZ * T_SZ * NUM_HEADS * NA_SZ * NA_SZ * 4);

  auto F  = [&](int i) { return reinterpret_cast<const float*>(d_in[i]); };
  auto wconv = [&](int src, int K, int N, bf16_t* dst) {
    convert_wT_kernel<<<(K * N) / 256, 256, 0, stream>>>(F(src), dst, K, N);
  };
  wconv(T_QW, 256, 256,  WqT_t); wconv(T_KW, 256, 256,  WkT_t);
  wconv(T_VW, 256, 256,  WvT_t); wconv(T_OW, 256, 256,  WoT_t);
  wconv(TF_W1, 256, 1024, W1T_t); wconv(TF_W2, 1024, 256, W2T_t);
  wconv(A_QW, 256, 256,  WqT_a); wconv(A_KW, 256, 256,  WkT_a);
  wconv(A_VW, 256, 256,  WvT_a); wconv(A_OW, 256, 256,  WoT_a);
  wconv(AF_W1, 256, 1024, W1T_a); wconv(AF_W2, 1024, 256, W2T_a);
  wconv(Q_L1W, 256, 128, L1T);   wconv(Q_L2W, 128, 128, L2T);

  auto gemm = [&](const bf16_t* A, const bf16_t* Wt, const float* bias, void* C,
                  int Mm, int Nn, int Kk, int mode) {
    dim3 g(Nn / 128, Mm / 64);
    gemm_kernel<<<g, 256, 0, stream>>>(A, Wt, bias, C, Mm, Nn, Kk, mode);
  };
  const float scale = 0.17677669529663687f;   // 1/sqrt(32)
  const int LN_GRID = MTOK / 8;

  // ---- temporal stage (rows: (b*NA+a)*T + t) ----
  prep_x_kernel<<<MTOK, 256, 0, stream>>>(F(IN_X), xa_bf, xa_f);
  gemm(xa_bf, WqT_t, F(T_QB), qb, MTOK, 256, 256, 2);
  gemm(xa_bf, WkT_t, F(T_KB), kb, MTOK, 256, 256, 2);
  gemm(xa_bf, WvT_t, F(T_VB), vb, MTOK, 256, 256, 2);
  attn_kernel<4><<<(B_SZ * NA_SZ) * NUM_HEADS, 128, 0, stream>>>(
      qb, kb, vb, F(IN_TMASK), T_SZ * T_SZ, 0, ao, scale);
  gemm(ao, WoT_t, F(T_OB), gout, MTOK, 256, 256, 0);
  add_ln_kernel<<<LN_GRID, 256, 0, stream>>>(xa_f, gout, F(T_LNG), F(T_LNB),
                                             xb_bf, xb_f, 0);
  gemm(xb_bf, W1T_t, F(TF_B1), hbuf, MTOK, 1024, 256, 1);
  gemm(hbuf, W2T_t, F(TF_B2), gout, MTOK, 256, 1024, 0);
  // LN + permute to asset layout (rows: (b*T+t)*NA + a) -> pair A
  add_ln_kernel<<<LN_GRID, 256, 0, stream>>>(xb_f, gout, F(TF_LNG), F(TF_LNB),
                                             xa_bf, xa_f, 1);

  // ---- factored dynamic bias ----
  gemm(xa_bf, L1T, F(Q_L1B), q1, MTOK, 128, 256, 1);
  gemm(q1, L2T, nullptr, qvec, MTOK, 128, 128, 0);
  dyn_bias_kernel<<<MTOK, 256, 0, stream>>>(qvec, F(IN_BETA), F(IN_GAMMA), abias);

  // ---- asset stage (rows: (b*T+t)*NA + a) ----
  gemm(xa_bf, WqT_a, F(A_QB), qb, MTOK, 256, 256, 2);
  gemm(xa_bf, WkT_a, F(A_KB), kb, MTOK, 256, 256, 2);
  gemm(xa_bf, WvT_a, F(A_VB), vb, MTOK, 256, 256, 2);
  attn_kernel<2><<<(B_SZ * T_SZ) * NUM_HEADS, 64, 0, stream>>>(
      qb, kb, vb, abias, NUM_HEADS * NA_SZ * NA_SZ, NA_SZ * NA_SZ, ao, scale);
  gemm(ao, WoT_a, F(A_OB), gout, MTOK, 256, 256, 0);
  add_ln_kernel<<<LN_GRID, 256, 0, stream>>>(xa_f, gout, F(A_LNG), F(A_LNB),
                                             xb_bf, xb_f, 0);
  gemm(xb_bf, W1T_a, F(AF_B1), hbuf, MTOK, 1024, 256, 1);
  gemm(hbuf, W2T_a, F(AF_B2), gout, MTOK, 256, 1024, 0);
  // final LN writes fp32 output directly (asset layout == [B,T,NA,D])
  add_ln_kernel<<<LN_GRID, 256, 0, stream>>>(xb_f, gout, F(AF_LNG), F(AF_LNB),
                                             nullptr, reinterpret_cast<float*>(d_out), 0);
}
